// MambaPlusPlus_optimized_88596585382317
// MI455X (gfx1250) — compile-verified, matrix operand order checked by reading, not run on hardware
//
#include <hip/hip_runtime.h>
#include <hip/hip_bf16.h>
#include <math.h>

// ---------------------------------------------------------------------------
// Problem constants (from the reference): B=2, L=1024, V=32000, E=1024,
// H=16, D=64, HID=1024.  M = B*L = 2048 rows everywhere.
// ---------------------------------------------------------------------------
#define MB_M    2048
#define MB_E    1024
#define MB_HID  1024
#define MB_V    32000
#define MB_H    16
#define MB_D    64
#define MB_FFN  4096
#define MB_EPS  1e-5f

typedef __bf16 bf16;
typedef __attribute__((ext_vector_type(16))) __bf16 v16bf;
typedef __attribute__((ext_vector_type(8)))  __bf16 v8bf;
typedef __attribute__((ext_vector_type(8)))  float  v8f;

// ---------------------------------------------------------------------------
// Fragment loader: row of 32 contiguous bf16 -> v16bf per the CDNA5 16-bit
// A/B 16x32 layout: lanes 0-15 hold K = kb..kb+7 (elems 0..7) and
// kb+16..kb+23 (elems 8..15) with kb = (lane>>4)*8.
// ---------------------------------------------------------------------------
__device__ __forceinline__ v16bf load_frag32(const bf16* __restrict__ row, int kb) {
    v8bf lo = *(const v8bf*)(row + kb);
    v8bf hi = *(const v8bf*)(row + kb + 16);
    v16bf f;
#pragma unroll
    for (int i = 0; i < 8; ++i) { f[i] = lo[i]; f[i + 8] = hi[i]; }
    return f;
}

#define WMMA_BF16(a, b, c) \
    __builtin_amdgcn_wmma_f32_16x16x32_bf16(false, (a), false, (b), (short)0, (c), false, false)

// ---------------------------------------------------------------------------
// Generic bf16 GEMM, f32 accumulate:  C[M,N] = A[M,K] * B[K,N] (+ epilogue)
// MODE 0: f32 out  = acc + bias[n]
// MODE 1: bf16 out = acc + bias[n]
// MODE 2: bf16 out = gelu(acc + bias[n])        (exact erf gelu)
// MODE 3: bf16 out = acc + bias[n] + res[m,n]   (res is f32)
//
// Block: 256 threads (8 waves); block tile 128x64; waves arranged 4(M)x2(N),
// each wave owns a 32x32 region = four 16x16 WMMA accumulators (2 A-frags x
// 2 B-frags -> 4 WMMAs per K-step).  Ping-pong LDS double buffering with
// register-staged global loads to overlap VMEM with the matrix pipe.
// ---------------------------------------------------------------------------
template <int MODE>
__global__ __launch_bounds__(256)
void gemm_bf16_wmma(const bf16* __restrict__ A, const bf16* __restrict__ B,
                    const float* __restrict__ bias, const float* __restrict__ res,
                    void* __restrict__ Cout, int M, int N, int K)
{
    __shared__ bf16 As[2][128][32];
    __shared__ bf16 Bs[2][64][32];   // transposed: Bs[buf][n][k]

    const int m0 = blockIdx.y * 128;
    const int n0 = blockIdx.x * 64;

    const int tid  = threadIdx.x;
    const int wave = tid >> 5;
    const int lane = tid & 31;
    const int wm   = (wave & 3) * 32;   // wave M offset within block tile
    const int wn   = (wave >> 2) * 32;  // wave N offset within block tile

    v8f acc[2][2] = {};

    // staging indices (256 threads cover 128x32 A-tile in two halves + 32x64 B-tile)
    const int ar = tid >> 2;            // 0..63  A row (and +64)
    const int ac = (tid & 3) * 8;       // col group of 8 bf16
    const int bk = tid >> 3;            // 0..31  B row (K)
    const int bn = (tid & 7) * 8;       // col group of 8 bf16

    uint4 ra0, ra1, rb;
    auto gload = [&](int k0) {
        const bf16* sa0 = A + (size_t)(m0 + ar) * K + (k0 + ac);
        const bf16* sa1 = A + (size_t)(m0 + 64 + ar) * K + (k0 + ac);
        const bf16* sb  = B + (size_t)(k0 + bk) * N + (n0 + bn);
        ra0 = *(const uint4*)sa0;
        ra1 = *(const uint4*)sa1;
        rb  = *(const uint4*)sb;
        if (k0 + 32 < K) {
            __builtin_prefetch(sa0 + 32, 0, 1);
            __builtin_prefetch(sb + (size_t)32 * N, 0, 1);
        }
    };
    auto sstore = [&](int buf) {
        *(uint4*)&As[buf][ar][ac]      = ra0;
        *(uint4*)&As[buf][64 + ar][ac] = ra1;
        const bf16* e = (const bf16*)&rb;
#pragma unroll
        for (int i = 0; i < 8; ++i) Bs[buf][bn + i][bk] = e[i];
    };

    gload(0);
    sstore(0);
    __syncthreads();

    int buf = 0;
    for (int k0 = 0; k0 < K; k0 += 32) {
        const bool more = (k0 + 32 < K);
        if (more) gload(k0 + 32);

        const int kb  = (lane >> 4) * 8;
        const int l15 = lane & 15;
        v16bf af0 = load_frag32(&As[buf][wm + l15][0], kb);
        v16bf af1 = load_frag32(&As[buf][wm + 16 + l15][0], kb);
        v16bf bf0 = load_frag32(&Bs[buf][wn + l15][0], kb);
        v16bf bf1 = load_frag32(&Bs[buf][wn + 16 + l15][0], kb);

        acc[0][0] = WMMA_BF16(af0, bf0, acc[0][0]);
        acc[0][1] = WMMA_BF16(af0, bf1, acc[0][1]);
        acc[1][0] = WMMA_BF16(af1, bf0, acc[1][0]);
        acc[1][1] = WMMA_BF16(af1, bf1, acc[1][1]);

        if (more) sstore(buf ^ 1);
        __syncthreads();
        buf ^= 1;
    }

    // ---- epilogue: C/D layout -> row = r + (lane>>4)*8, col = lane&15 ----
    const int l15  = lane & 15;
    const int rsel = (lane >> 4) * 8;
#pragma unroll
    for (int i = 0; i < 2; ++i) {
#pragma unroll
        for (int j = 0; j < 2; ++j) {
            const int n = n0 + wn + j * 16 + l15;
            const float bn_ = bias ? bias[n] : 0.0f;
#pragma unroll
            for (int r = 0; r < 8; ++r) {
                const int m = m0 + wm + i * 16 + rsel + r;
                float v = acc[i][j][r] + bn_;
                const size_t idx = (size_t)m * N + n;
                if constexpr (MODE == 0) {
                    ((float*)Cout)[idx] = v;
                } else if constexpr (MODE == 1) {
                    ((bf16*)Cout)[idx] = (bf16)v;
                } else if constexpr (MODE == 2) {
                    v = 0.5f * v * (1.0f + erff(v * 0.70710678118654752f));
                    ((bf16*)Cout)[idx] = (bf16)v;
                } else { // MODE 3
                    v += res[idx];
                    ((bf16*)Cout)[idx] = (bf16)v;
                }
            }
        }
    }
}

// ---------------------------------------------------------------------------
// Block-diagonal per-head C projection + gating:
//   z[m, h*64+e] = (sum_d hs[m, h*64+d] * Cw[h,d,e] + Cb[h*64+e]) * wout[m, h*64+e]
// cwT is Cw pre-transposed: cwT[h][e][d].  8 waves per block; wave w handles
// column tile (blockIdx.x*8 + w), 16 rows per block row.  K=64 (2 WMMA steps).
// wout lives in abw at column offset 2048, ld = 3072.
// ---------------------------------------------------------------------------
__global__ __launch_bounds__(256)
void cproj_wmma(const bf16* __restrict__ hs, const bf16* __restrict__ cwT,
                const float* __restrict__ Cb, const float* __restrict__ abw,
                float* __restrict__ z)
{
    const int wave  = threadIdx.x >> 5;
    const int ntile = blockIdx.x * 8 + wave;   // 0..63
    const int m0    = blockIdx.y * 16;         // 0..2032
    const int h     = ntile >> 2;
    const int np    = (ntile & 3) * 16;        // col within head
    const int lane  = threadIdx.x & 31;
    const int l15   = lane & 15;
    const int kb    = (lane >> 4) * 8;

    v8f acc = {};
#pragma unroll
    for (int ks = 0; ks < 64; ks += 32) {
        const bf16* arow = hs + (size_t)(m0 + l15) * MB_HID + h * 64 + ks;
        const bf16* brow = cwT + (size_t)h * 4096 + (size_t)(np + l15) * 64 + ks;
        v16bf af = load_frag32(arow, kb);
        v16bf bf = load_frag32(brow, kb);
        acc = WMMA_BF16(af, bf, acc);
    }

    const int col  = h * 64 + np + l15;
    const int row0 = m0 + (lane >> 4) * 8;
    const float cb = Cb[col];
#pragma unroll
    for (int r = 0; r < 8; ++r) {
        const int m = row0 + r;
        float v = acc[r] + cb;
        v *= abw[(size_t)m * 3072 + 2048 + col];
        z[(size_t)m * MB_HID + col] = v;
    }
}

// ---------------------------------------------------------------------------
// Sequential recurrence: h = tanh(a)*h + b*b, per (batch, col) state.
// 2048 independent states, each walks L=1024 steps. Coalesced over col.
// ---------------------------------------------------------------------------
__global__ __launch_bounds__(256)
void scan_kernel(const float* __restrict__ abw, bf16* __restrict__ hs)
{
    const int t   = blockIdx.x * blockDim.x + threadIdx.x;   // 0..2047
    const int b   = t >> 10;
    const int col = t & 1023;
    float h = 0.0f;
    for (int l = 0; l < 1024; ++l) {
        const size_t row = (size_t)(b * 1024 + l);
        const float a  = tanhf(abw[row * 3072 + col]);
        const float bv = abw[row * 3072 + 1024 + col];
        h = a * h + bv * bv;
        hs[row * MB_HID + col] = (bf16)h;
    }
}

// ---------------------------------------------------------------------------
// u = z + LayerNorm(z); writes u as f32 (residual) and bf16 (GEMM input).
// One 256-thread block per row of 1024.
// ---------------------------------------------------------------------------
__global__ __launch_bounds__(256)
void ln_residual_kernel(const float* __restrict__ z, const float* __restrict__ gamma,
                        const float* __restrict__ beta, float* __restrict__ u,
                        bf16* __restrict__ ubf)
{
    __shared__ float s1[256], s2[256];
    const int m = blockIdx.x;
    const float* zr = z + (size_t)m * MB_HID;
    float ls = 0.0f, lq = 0.0f;
    for (int j = threadIdx.x; j < MB_HID; j += 256) {
        float v = zr[j];
        ls += v; lq += v * v;
    }
    s1[threadIdx.x] = ls; s2[threadIdx.x] = lq;
    __syncthreads();
    for (int off = 128; off > 0; off >>= 1) {
        if (threadIdx.x < off) {
            s1[threadIdx.x] += s1[threadIdx.x + off];
            s2[threadIdx.x] += s2[threadIdx.x + off];
        }
        __syncthreads();
    }
    const float mean = s1[0] * (1.0f / MB_HID);
    const float var  = s2[0] * (1.0f / MB_HID) - mean * mean;
    const float rstd = rsqrtf(var + MB_EPS);
    for (int j = threadIdx.x; j < MB_HID; j += 256) {
        float v  = zr[j];
        float ln = (v - mean) * rstd * gamma[j] + beta[j];
        float uu = v + ln;
        u[(size_t)m * MB_HID + j]   = uu;
        ubf[(size_t)m * MB_HID + j] = (bf16)uu;
    }
}

// ---------------------------------------------------------------------------
// Small prep kernels
// ---------------------------------------------------------------------------
__global__ void embed_gather_kernel(const int* __restrict__ x,
                                    const float* __restrict__ embed,
                                    bf16* __restrict__ out)
{
    const int m = blockIdx.x;                  // 0..2047
    const int tok = x[m];
    const float* src = embed + (size_t)tok * MB_E;
    bf16* dst = out + (size_t)m * MB_E;
    for (int j = threadIdx.x; j < MB_E; j += blockDim.x)
        dst[j] = (bf16)src[j];
}

__global__ void convert_f32_bf16_kernel(const float* __restrict__ in,
                                        bf16* __restrict__ out, size_t n)
{
    for (size_t i = (size_t)blockIdx.x * blockDim.x + threadIdx.x; i < n;
         i += (size_t)gridDim.x * blockDim.x)
        out[i] = (bf16)in[i];
}

// Pack Wa|Wb|Wo [H,E,D] -> B matrix [E, 3*HID]: out[e*3072 + p*1024 + h*64+d]
__global__ void pack_w3_kernel(const float* __restrict__ Wa,
                               const float* __restrict__ Wb,
                               const float* __restrict__ Wo,
                               bf16* __restrict__ out)
{
    const size_t total = (size_t)3 * MB_E * MB_HID;
    for (size_t i = (size_t)blockIdx.x * blockDim.x + threadIdx.x; i < total;
         i += (size_t)gridDim.x * blockDim.x) {
        const int e = (int)(i / 3072);
        const int c = (int)(i % 3072);
        const int p = c >> 10;              // which weight
        const int hd = c & 1023;            // h*64+d
        const int h = hd >> 6, d = hd & 63;
        const float* W = (p == 0) ? Wa : (p == 1) ? Wb : Wo;
        out[i] = (bf16)W[(size_t)h * MB_E * MB_D + (size_t)e * MB_D + d];
    }
}

// Transpose Cw [H,D,E'] -> cwT [H,E',D] (bf16)
__global__ void pack_cwT_kernel(const float* __restrict__ Cw, bf16* __restrict__ out)
{
    const int total = MB_H * MB_D * MB_D;   // 65536
    for (int i = blockIdx.x * blockDim.x + threadIdx.x; i < total;
         i += gridDim.x * blockDim.x) {
        const int h = i >> 12, e = (i >> 6) & 63, d = i & 63;
        out[i] = (bf16)Cw[(size_t)h * 4096 + (size_t)d * 64 + e];
    }
}

__global__ void concat3_kernel(const float* __restrict__ a, const float* __restrict__ b,
                               const float* __restrict__ c, float* __restrict__ out)
{
    const int i = blockIdx.x * blockDim.x + threadIdx.x;   // 0..3071
    if (i < 1024)       out[i] = a[i];
    else if (i < 2048)  out[i] = b[i - 1024];
    else if (i < 3072)  out[i] = c[i - 2048];
}

// ---------------------------------------------------------------------------
// Host side
// ---------------------------------------------------------------------------
extern "C" void kernel_launch(void* const* d_in, const int* in_sizes, int n_in,
                              void* d_out, int out_size, void* d_ws, size_t ws_size,
                              hipStream_t stream)
{
    (void)in_sizes; (void)n_in; (void)out_size; (void)ws_size;
    const int*   x       = (const int*)  d_in[0];
    const float* embed   = (const float*)d_in[1];
    const float* Wa      = (const float*)d_in[2];
    const float* ba      = (const float*)d_in[3];
    const float* Wb      = (const float*)d_in[4];
    const float* bb      = (const float*)d_in[5];
    const float* Wo      = (const float*)d_in[6];
    const float* bo      = (const float*)d_in[7];
    const float* Cw      = (const float*)d_in[8];
    const float* Cb      = (const float*)d_in[9];
    const float* proj_w  = (const float*)d_in[10];
    const float* proj_b  = (const float*)d_in[11];
    const float* gamma   = (const float*)d_in[12];
    const float* beta    = (const float*)d_in[13];
    const float* ffn1_w  = (const float*)d_in[14];
    const float* ffn1_b  = (const float*)d_in[15];
    const float* ffn2_w  = (const float*)d_in[16];
    const float* ffn2_b  = (const float*)d_in[17];
    const float* out_w   = (const float*)d_in[18];
    const float* out_b   = (const float*)d_in[19];
    float* logits = (float*)d_out;

    // Workspace carving (256B aligned slices)
    char* wsp = (char*)d_ws;
    auto alloc = [&](size_t bytes) -> void* {
        void* p = (void*)wsp;
        wsp += (bytes + 255) & ~(size_t)255;
        return p;
    };
    bf16*  emb_bf   = (bf16*) alloc((size_t)MB_M * MB_E * 2);
    bf16*  w_abo    = (bf16*) alloc((size_t)MB_E * 3072 * 2);
    float* bias_abw = (float*)alloc(3072 * 4);
    bf16*  cwT      = (bf16*) alloc((size_t)MB_H * 4096 * 2);
    bf16*  projw_bf = (bf16*) alloc((size_t)MB_HID * MB_HID * 2);
    bf16*  ffn1w_bf = (bf16*) alloc((size_t)MB_HID * MB_FFN * 2);
    bf16*  ffn2w_bf = (bf16*) alloc((size_t)MB_FFN * MB_HID * 2);
    bf16*  outw_bf  = (bf16*) alloc((size_t)MB_HID * MB_V * 2);
    float* abw      = (float*)alloc((size_t)MB_M * 3072 * 4);
    bf16*  hs       = (bf16*) alloc((size_t)MB_M * MB_HID * 2);
    float* z        = (float*)alloc((size_t)MB_M * MB_HID * 4);
    float* u_f32    = (float*)alloc((size_t)MB_M * MB_HID * 4);
    bf16*  u_bf     = (bf16*) alloc((size_t)MB_M * MB_HID * 2);
    bf16*  act_bf   = (bf16*) alloc((size_t)MB_M * MB_FFN * 2);
    bf16*  hout_bf  = (bf16*) alloc((size_t)MB_M * MB_HID * 2);
    bf16*  t_bf     = (bf16*) alloc((size_t)MB_M * MB_HID * 2);

    // --- prep: gather + weight conversion ---
    embed_gather_kernel<<<MB_M, 256, 0, stream>>>(x, embed, emb_bf);
    pack_w3_kernel<<<2048, 256, 0, stream>>>(Wa, Wb, Wo, w_abo);
    concat3_kernel<<<12, 256, 0, stream>>>(ba, bb, bo, bias_abw);
    pack_cwT_kernel<<<64, 256, 0, stream>>>(Cw, cwT);
    convert_f32_bf16_kernel<<<2048, 256, 0, stream>>>(proj_w, projw_bf,
                                                      (size_t)MB_HID * MB_HID);
    convert_f32_bf16_kernel<<<4096, 256, 0, stream>>>(ffn1_w, ffn1w_bf,
                                                      (size_t)MB_HID * MB_FFN);
    convert_f32_bf16_kernel<<<4096, 256, 0, stream>>>(ffn2_w, ffn2w_bf,
                                                      (size_t)MB_FFN * MB_HID);
    convert_f32_bf16_kernel<<<8192, 256, 0, stream>>>(out_w, outw_bf,
                                                      (size_t)MB_HID * MB_V);

    // --- fused a/b/wout projections: [2048,1024] x [1024,3072] -> f32 ---
    gemm_bf16_wmma<0><<<dim3(3072 / 64, MB_M / 128), 256, 0, stream>>>(
        emb_bf, w_abo, bias_abw, nullptr, abw, MB_M, 3072, MB_E);

    // --- recurrence (tanh applied to a inside) ---
    scan_kernel<<<8, 256, 0, stream>>>(abw, hs);

    // --- per-head C projection + output gating -> z ---
    cproj_wmma<<<dim3(8, MB_M / 16), 256, 0, stream>>>(hs, cwT, Cb, abw, z);

    // --- u = z + LN(z) ---
    ln_residual_kernel<<<MB_M, 256, 0, stream>>>(z, gamma, beta, u_f32, u_bf);

    // --- FFN1 + exact GELU -> bf16 activations ---
    gemm_bf16_wmma<2><<<dim3(MB_FFN / 64, MB_M / 128), 256, 0, stream>>>(
        u_bf, ffn1w_bf, ffn1_b, nullptr, act_bf, MB_M, MB_FFN, MB_HID);

    // --- FFN2 + residual(u) -> h_out (bf16) ---
    gemm_bf16_wmma<3><<<dim3(MB_HID / 64, MB_M / 128), 256, 0, stream>>>(
        act_bf, ffn2w_bf, ffn2_b, u_f32, hout_bf, MB_M, MB_HID, MB_FFN);

    // --- proj GEMM -> t (bf16) ---
    gemm_bf16_wmma<1><<<dim3(MB_HID / 64, MB_M / 128), 256, 0, stream>>>(
        hout_bf, projw_bf, proj_b, nullptr, t_bf, MB_M, MB_HID, MB_HID);

    // --- vocab GEMM -> logits (f32, dominant cost) ---
    gemm_bf16_wmma<0><<<dim3(MB_V / 64, MB_M / 128), 256, 0, stream>>>(
        t_bf, outw_bf, out_b, nullptr, logits, MB_M, MB_V, MB_HID);
}
